// WaveFieldAttentionWavelet_46024869544530
// MI455X (gfx1250) — matrix operand order repaired and checked by reading, not compiled, not run on hardware
//
#include <hip/hip_runtime.h>

#define B_ 4
#define N_ 2048
#define D_ 1024
#define H_ 16
#define HD_ 64
#define G_ 4096
#define LVLS 12
#define M_ (B_*N_)          // 8192 rows
#define FFT_N 8192
#define PI_F 3.14159265358979323846f

typedef float v8f __attribute__((ext_vector_type(8)));
typedef __bf16 bf16;
typedef bf16 v16bf __attribute__((ext_vector_type(16)));

// ---------------------------------------------------------------------------
// GEMM: Y[M][N] = X[M][K] * W[N][K]^T + bias[N]   (f32 in, bf16 WMMA, f32 acc)
// Block: 256 threads = 8 waves; tile 128x128, BK=32.
// Wave (wm,wn) computes 64x32 via 4x2 array of 16x16 wmma accumulators.
// Global->LDS staging is software-pipelined through registers so the next
// tile's loads are in flight while the current tile's WMMAs execute.
// ---------------------------------------------------------------------------
#define BM 128
#define BN 128
#define BK 32
#define LDT 48   // padded LDS row stride (bf16 elems)

__global__ __launch_bounds__(256) void gemm_xwt(const float* __restrict__ X,
                                                const float* __restrict__ W,
                                                const float* __restrict__ bias,
                                                float* __restrict__ Y,
                                                int M, int N, int K)
{
  __shared__ __align__(16) bf16 As[BM][LDT];
  __shared__ __align__(16) bf16 Bs[BN][LDT];
  const int tid  = threadIdx.x;
  const int lane = tid & 31;
  const int wv   = tid >> 5;
  const int wm   = wv & 1;      // 0..1 -> 64-row half
  const int wn   = wv >> 1;     // 0..3 -> 32-col quarter
  const int g    = lane >> 4;   // lane group (0/1)
  const int ln   = lane & 15;
  const int bm   = blockIdx.y * BM;
  const int bn   = blockIdx.x * BN;

  // per-thread tile-load geometry: 4 float4 per matrix per tile
  const int lr = tid >> 1;              // unused helper removed below
  (void)lr;

  v8f acc[4][2] = {};
  float4 ra[4], rb[4];

  // ---- prologue: load tile kc=0 into registers (one load clause) ----
#pragma unroll
  for (int i = 0; i < 4; i++) {
    int id = tid + i * 256;             // float4 index, 8 per row
    int r  = id >> 3;
    int c  = (id & 7) << 2;
    ra[i] = *(const float4*)(X + (size_t)(bm + r) * K + c);
    rb[i] = *(const float4*)(W + (size_t)(bn + r) * K + c);
  }

  for (int kc = 0; kc < K; kc += BK) {
    // ---- stage current registers into LDS (cvt f32 -> bf16) ----
#pragma unroll
    for (int i = 0; i < 4; i++) {
      int id = tid + i * 256;
      int r  = id >> 3;
      int c  = (id & 7) << 2;
      As[r][c+0] = (bf16)ra[i].x; As[r][c+1] = (bf16)ra[i].y;
      As[r][c+2] = (bf16)ra[i].z; As[r][c+3] = (bf16)ra[i].w;
      Bs[r][c+0] = (bf16)rb[i].x; Bs[r][c+1] = (bf16)rb[i].y;
      Bs[r][c+2] = (bf16)rb[i].z; Bs[r][c+3] = (bf16)rb[i].w;
    }
    __syncthreads();

    // ---- issue next tile's global loads (overlap with WMMA below) ----
    if (kc + BK < K) {
#pragma unroll
      for (int i = 0; i < 4; i++) {
        int id = tid + i * 256;
        int r  = id >> 3;
        int c  = (id & 7) << 2;
        ra[i] = *(const float4*)(X + (size_t)(bm + r) * K + kc + BK + c);
        rb[i] = *(const float4*)(W + (size_t)(bn + r) * K + kc + BK + c);
      }
    }
    if (kc + 2 * BK < K) {   // gfx1250 global_prefetch_b8 two tiles ahead
      int r = tid & 127;
      __builtin_prefetch(X + (size_t)(bm + r) * K + kc + 2 * BK, 0, 1);
      __builtin_prefetch(W + (size_t)(bn + r) * K + kc + 2 * BK, 0, 1);
    }

    // ---- fragments per ISA VGPR layouts (16-bit A 16x32; B 32x16) ----
    v16bf af[4];
    v16bf bfr[2];
#pragma unroll
    for (int mi = 0; mi < 4; mi++) {
      int m = wm * 64 + mi * 16 + ln;
#pragma unroll
      for (int e = 0; e < 8; e++) {
        af[mi][e]     = As[m][g*8 + e];        // K = g*8 + e
        af[mi][8 + e] = As[m][16 + g*8 + e];   // K = 16 + g*8 + e
      }
    }
#pragma unroll
    for (int ni = 0; ni < 2; ni++) {
      int n = wn * 32 + ni * 16 + ln;
#pragma unroll
      for (int e = 0; e < 16; e++)
        bfr[ni][e] = Bs[n][(e >> 3) * 16 + g*8 + (e & 7)];  // K = (e/8)*16 + g*8 + e%8
    }
#pragma unroll
    for (int mi = 0; mi < 4; mi++)
#pragma unroll
      for (int ni = 0; ni < 2; ni++)
        acc[mi][ni] = __builtin_amdgcn_wmma_f32_16x16x32_bf16(
            false, af[mi], false, bfr[ni], (short)0, acc[mi][ni], false, false);
    __syncthreads();
  }

  // Epilogue: C/D layout => VGPR v, lanes 0-15 -> M=v, lanes 16-31 -> M=v+8
#pragma unroll
  for (int mi = 0; mi < 4; mi++) {
#pragma unroll
    for (int ni = 0; ni < 2; ni++) {
      int mrow = bm + wm*64 + mi*16 + g*8;
      int ncol = bn + wn*32 + ni*16 + ln;
      float bv = bias[ncol];
#pragma unroll
      for (int v = 0; v < 8; v++)
        Y[(size_t)(mrow + v) * N + ncol] = acc[mi][ni][v] + bv;
    }
  }
}

// ---------------------------------------------------------------------------
// In-LDS radix-2 complex FFT, length 8192, 256 threads. sign=-1 fwd, +1 inv.
// ---------------------------------------------------------------------------
__device__ void fft8192(float2* s, float sign)
{
  const int tid = threadIdx.x;
  // bit reversal (13 bits)
  for (int i = tid; i < FFT_N; i += 256) {
    int j = (int)(__brev((unsigned)i) >> 19);
    if (j > i) { float2 t = s[i]; s[i] = s[j]; s[j] = t; }
  }
  __syncthreads();
  for (int stage = 1; stage <= 13; stage++) {
    const int half = 1 << (stage - 1);
    for (int bu = tid; bu < FFT_N / 2; bu += 256) {
      int grp = bu >> (stage - 1);
      int pos = bu & (half - 1);
      int i = (grp << stage) + pos;
      int j = i + half;
      float ang = sign * PI_F * (float)pos / (float)half;
      float sn, cs;
      __sincosf(ang, &sn, &cs);
      float2 u = s[i];
      float2 t = s[j];
      float2 tw = make_float2(t.x * cs - t.y * sn, t.x * sn + t.y * cs);
      s[i] = make_float2(u.x + tw.x, u.y + tw.y);
      s[j] = make_float2(u.x - tw.x, u.y - tw.y);
    }
    __syncthreads();
  }
}

// ---------------------------------------------------------------------------
// Build per-head wavelet kernel (time domain, normalized) and its 8192-pt FFT.
// One block per head. LDS buffer reused as scratch (f32 view) then complex.
// ---------------------------------------------------------------------------
__global__ __launch_bounds__(256) void build_kfft(const float* __restrict__ gains,
                                                  float2* __restrict__ KFFT)
{
  __shared__ float2 s[FFT_N];          // 64 KB
  float* f = (float*)s;                // 16384 f32 view
  const int h = blockIdx.x;
  const int tid = threadIdx.x;

  // per-level sum of |wav| over t (each level reduced across block)
  float wsum[LVLS];
  for (int lv = 0; lv < LVLS; lv++) {
    float sig = exp2f(11.0f - (float)lv);
    float local = 0.0f;
    for (int t = tid; t < G_; t += 256) {
      float ts = (float)t / sig;
      float w = __expf(-0.5f * ts * ts) * __cosf(PI_F * ts);
      local += fabsf(w);
    }
    f[FFT_N + tid] = local;
    __syncthreads();
    for (int off = 128; off > 0; off >>= 1) {
      if (tid < off) f[FFT_N + tid] += f[FFT_N + tid + off];
      __syncthreads();
    }
    if (tid == 0) f[FFT_N + 512 + lv] = fmaxf(f[FFT_N], 1e-8f);
    __syncthreads();
  }
  for (int lv = 0; lv < LVLS; lv++) wsum[lv] = f[FFT_N + 512 + lv];
  __syncthreads();

  // kernel row = sum_s gains[h,s] * wav_norm(t,s); also accumulate |kernel|
  float local = 0.0f;
  for (int t = tid; t < G_; t += 256) {
    float kv = 0.0f;
#pragma unroll
    for (int lv = 0; lv < LVLS; lv++) {
      float sig = exp2f(11.0f - (float)lv);
      float ts = (float)t / sig;
      float w = __expf(-0.5f * ts * ts) * __cosf(PI_F * ts);
      kv += gains[h * LVLS + lv] * (w / wsum[lv]);
    }
    f[t] = kv;
    local += fabsf(kv);
  }
  f[FFT_N + tid] = local;
  __syncthreads();
  for (int off = 128; off > 0; off >>= 1) {
    if (tid < off) f[FFT_N + tid] += f[FFT_N + tid + off];
    __syncthreads();
  }
  if (tid == 0) f[FFT_N + 512] = fmaxf(f[FFT_N], 1e-8f);
  __syncthreads();
  float invk = 1.0f / f[FFT_N + 512];

  // move normalized kernel into complex buffer (regs first, f aliases s)
  float regs[16];
#pragma unroll
  for (int j = 0; j < 16; j++) regs[j] = f[tid + j * 256] * invk;
  __syncthreads();
#pragma unroll
  for (int j = 0; j < 16; j++) {
    int t = tid + j * 256;
    s[t] = make_float2(regs[j], 0.0f);
    s[t + G_] = make_float2(0.0f, 0.0f);
  }
  __syncthreads();

  fft8192(s, -1.0f);

  for (int i = tid; i < FFT_N; i += 256)
    KFFT[(size_t)h * FFT_N + i] = s[i];
}

// ---------------------------------------------------------------------------
// Deposit: field[b][h][hd][g] += v*||k|| with linear interp weights (atomics).
// One block per (b,n); 256 threads cover 4 heads x 64 hd per pass, 4 passes.
// ---------------------------------------------------------------------------
__global__ __launch_bounds__(256) void deposit_field(const float* __restrict__ KBUF,
                                                     const float* __restrict__ VBUF,
                                                     float* __restrict__ FIELD)
{
  __shared__ float sh[256];
  __shared__ float kmr[4];
  const int tid = threadIdx.x;
  const int b = blockIdx.x >> 11;
  const int n = blockIdx.x & (N_ - 1);

  const float stride = (float)(4095.0 / 2047.0);
  float pos = fminf((float)n * stride, (float)(G_ - 2));
  int lo = (int)pos; if (lo > G_ - 2) lo = G_ - 2;
  float frac = fminf(fmaxf(pos - (float)lo, 0.0f), 1.0f);
  const float wlo = 1.0f - frac, whi = frac;
  const size_t rowbase = ((size_t)(b * N_ + n)) * D_;

  for (int hrep = 0; hrep < 4; hrep++) {
    int h  = hrep * 4 + (tid >> 6);
    int hd = tid & 63;
    float kv = KBUF[rowbase + h * HD_ + hd];
    sh[tid] = kv * kv;
    __syncthreads();
    if ((tid & 63) == 0) {
      float ssum = 0.0f;
      for (int i = 0; i < 64; i++) ssum += sh[tid + i];
      kmr[tid >> 6] = sqrtf(ssum);
    }
    __syncthreads();
    float kmag = kmr[tid >> 6];
    float dep = VBUF[rowbase + h * HD_ + hd] * kmag;
    size_t fb = ((size_t)(b * H_ + h) * HD_ + hd) * G_;
    __hip_atomic_fetch_add(&FIELD[fb + lo], dep * wlo,
                           __ATOMIC_RELAXED, __HIP_MEMORY_SCOPE_AGENT);
    __hip_atomic_fetch_add(&FIELD[fb + lo + 1], dep * whi,
                           __ATOMIC_RELAXED, __HIP_MEMORY_SCOPE_AGENT);
    __syncthreads();
  }
}

// ---------------------------------------------------------------------------
// Per-column FFT convolution. Real kernel => pack 2 real columns as complex;
// one fwd FFT + spectrum mul + one inv FFT convolves both columns.
// One block per (b, h, hd-pair): 4*16*32 = 2048 blocks.
// ---------------------------------------------------------------------------
__global__ __launch_bounds__(256) void field_conv(const float* __restrict__ FIELD,
                                                  const float2* __restrict__ KFFT,
                                                  float* __restrict__ CONV)
{
  __shared__ float2 s[FFT_N];          // 64 KB
  const int tid = threadIdx.x;
  const int idx = blockIdx.x;
  const int dp = idx & 31;             // hd pair
  const int bh = idx >> 5;
  const int h  = bh & (H_ - 1);
  const int b  = bh >> 4;
  const size_t base0 = ((size_t)(b * H_ + h) * HD_ + dp * 2) * G_;
  const size_t base1 = base0 + G_;

  for (int g = tid; g < G_; g += 256) {
    s[g] = make_float2(FIELD[base0 + g], FIELD[base1 + g]);
    s[g + G_] = make_float2(0.0f, 0.0f);
  }
  __syncthreads();

  fft8192(s, -1.0f);

  const float2* kf = KFFT + (size_t)h * FFT_N;
  for (int i = tid; i < FFT_N; i += 256) {
    float2 z = s[i];
    float2 k = kf[i];
    s[i] = make_float2(z.x * k.x - z.y * k.y, z.x * k.y + z.y * k.x);
  }
  __syncthreads();

  fft8192(s, +1.0f);

  const float inv = 1.0f / (float)FFT_N;
  for (int g = tid; g < G_; g += 256) {
    CONV[base0 + g] = s[g].x * inv;
    CONV[base1 + g] = s[g].y * inv;
  }
}

// ---------------------------------------------------------------------------
// Row-softmax of the 16x16 coupling matrix.
// ---------------------------------------------------------------------------
__global__ void coup_softmax(const float* __restrict__ C, float* __restrict__ OUT)
{
  int r = threadIdx.x;
  if (r < H_) {
    float mx = -3.4e38f;
    for (int j = 0; j < H_; j++) mx = fmaxf(mx, C[r * H_ + j]);
    float e[H_];
    float ssum = 0.0f;
    for (int j = 0; j < H_; j++) { e[j] = __expf(C[r * H_ + j] - mx); ssum += e[j]; }
    float inv = 1.0f / ssum;
    for (int j = 0; j < H_; j++) OUT[r * H_ + j] = e[j] * inv;
  }
}

// ---------------------------------------------------------------------------
// Gather: head-coupling mix of conv field at interp positions, * sigmoid gate.
// One block per (b,n); output layout (b, n, h*HD+hd).
// ---------------------------------------------------------------------------
__global__ __launch_bounds__(256) void gather_gate(const float* __restrict__ CONV,
                                                   const float* __restrict__ GBUF,
                                                   const float* __restrict__ COUP,
                                                   float* __restrict__ YBUF)
{
  __shared__ float cs[H_ * H_];
  const int tid = threadIdx.x;
  const int b = blockIdx.x >> 11;
  const int n = blockIdx.x & (N_ - 1);
  cs[tid] = COUP[tid];                 // 256 == 16*16

  const float stride = (float)(4095.0 / 2047.0);
  float pos = fminf((float)n * stride, (float)(G_ - 2));
  int lo = (int)pos; if (lo > G_ - 2) lo = G_ - 2;
  float frac = fminf(fmaxf(pos - (float)lo, 0.0f), 1.0f);
  const float wlo = 1.0f - frac, whi = frac;
  __syncthreads();

  const size_t rowbase = ((size_t)(b * N_ + n)) * D_;
  for (int rep = 0; rep < 4; rep++) {
    int d  = tid + rep * 256;
    int h  = d >> 6;
    int hd = d & 63;
    float accv = 0.0f;
#pragma unroll
    for (int j = 0; j < H_; j++) {
      size_t base = ((size_t)(b * H_ + j) * HD_ + hd) * G_;
      accv += cs[h * H_ + j] * (wlo * CONV[base + lo] + whi * CONV[base + lo + 1]);
    }
    float gp = GBUF[rowbase + d];
    float gate = 1.0f / (1.0f + __expf(-gp));
    YBUF[rowbase + d] = accv * gate;
  }
}

// ---------------------------------------------------------------------------
// Host-side launcher
// ---------------------------------------------------------------------------
extern "C" void kernel_launch(void* const* d_in, const int* in_sizes, int n_in,
                              void* d_out, int out_size, void* d_ws, size_t ws_size,
                              hipStream_t stream)
{
  (void)in_sizes; (void)n_in; (void)out_size; (void)ws_size;
  const float* x     = (const float*)d_in[0];
  const float* Wqkv  = (const float*)d_in[1];
  const float* bqkv  = (const float*)d_in[2];
  const float* Wgate = (const float*)d_in[3];
  const float* bgate = (const float*)d_in[4];
  const float* Wout  = (const float*)d_in[5];
  const float* bout  = (const float*)d_in[6];
  const float* gains = (const float*)d_in[7];
  const float* coup  = (const float*)d_in[8];

  char* ws = (char*)d_ws;
  const size_t SZ  = (size_t)M_ * D_ * sizeof(float);            // 33.5 MB
  const size_t FSZ = (size_t)B_ * H_ * HD_ * G_ * sizeof(float); // 67 MB
  float*  KBUF  = (float*)(ws);
  float*  VBUF  = (float*)(ws + SZ);
  float*  GBUF  = (float*)(ws + 2 * SZ);
  float*  FIELD = (float*)(ws + 3 * SZ);
  float*  CONV  = (float*)(ws + 3 * SZ + FSZ);
  float2* KFFT  = (float2*)(ws + 3 * SZ + 2 * FSZ);
  float*  COUPS = (float*)(ws + 3 * SZ + 2 * FSZ + (size_t)H_ * FFT_N * sizeof(float2));
  float*  YBUF  = KBUF;   // KBUF dead after deposit; alias for gather output

  hipMemsetAsync(FIELD, 0, FSZ, stream);

  dim3 blk(256);
  dim3 gemm_grid(D_ / BN, M_ / BM);   // (8, 64)

  // k, v, gate projections (q is unused by the reference -> skipped)
  gemm_xwt<<<gemm_grid, blk, 0, stream>>>(x, Wqkv + (size_t)D_ * D_,     bqkv + D_,     KBUF, M_, D_, D_);
  gemm_xwt<<<gemm_grid, blk, 0, stream>>>(x, Wqkv + (size_t)2 * D_ * D_, bqkv + 2 * D_, VBUF, M_, D_, D_);
  gemm_xwt<<<gemm_grid, blk, 0, stream>>>(x, Wgate,                      bgate,         GBUF, M_, D_, D_);

  build_kfft<<<H_, blk, 0, stream>>>(gains, KFFT);
  deposit_field<<<B_ * N_, blk, 0, stream>>>(KBUF, VBUF, FIELD);
  field_conv<<<B_ * H_ * (HD_ / 2), blk, 0, stream>>>(FIELD, KFFT, CONV);
  coup_softmax<<<1, H_, 0, stream>>>(coup, COUPS);
  gather_gate<<<B_ * N_, blk, 0, stream>>>(CONV, GBUF, COUPS, YBUF);

  // final projection
  gemm_xwt<<<gemm_grid, blk, 0, stream>>>(YBUF, Wout, bout, (float*)d_out, M_, D_, D_);
}
